// VisionTransformer_35536559407606
// MI455X (gfx1250) — matrix operand order repaired and checked by reading, not compiled, not run on hardware
//
#include <hip/hip_runtime.h>
#include <math.h>

// ---------------------------------------------------------------------------
// BEiT-style ViT forward for MI455X (gfx1250, wave32, WMMA bf16 16x16x32)
// Round 2: LDS-staged, double-buffered GEMM (128x128x32 block tile).
// ---------------------------------------------------------------------------

typedef __bf16 bf16;
typedef __attribute__((ext_vector_type(16))) __bf16 v16bf;
typedef __attribute__((ext_vector_type(8)))  __bf16 v8bf;
typedef __attribute__((ext_vector_type(8)))  float  v8f;

#define Bb    64
#define Dd    768
#define Ll    12
#define Hh    12
#define HDm   64
#define FFm   3072
#define Nn    197
#define NP    224          // padded token count (multiple of 32)
#define NRDm  732
#define Mrows (Bb * Nn)    // 12608
#define MPp   (Bb * 196)   // 12544
#define SCL   0.125f       // 64^-0.5

#define BM 128
#define BN 128
#define BK 32
#define LDA 40             // LDS A row stride (elems), 80B -> spread bank groups
#define LDB 136            // LDS B row stride (elems), 272B

// ---- WMMA helpers ---------------------------------------------------------

__device__ __forceinline__ v8f wmma_bf16(v16bf a, v16bf b, v8f c) {
  return __builtin_amdgcn_wmma_f32_16x16x32_bf16(
      /*neg_a=*/false, a, /*neg_b=*/false, b,
      /*c_mod=*/(short)0, c, /*reuse_a=*/false, /*reuse_b=*/false);
}

// A-fragment: lane m = lane&15, K chunks [hi8..hi8+7] and [hi8+16..hi8+23].
__device__ __forceinline__ v16bf ld_frag_rowK(const bf16* p) {
  union { v16bf v; v8bf h[2]; } u;
  u.h[0] = *(const v8bf*)(p);
  u.h[1] = *(const v8bf*)(p + 16);
  return u.v;
}

// B-fragment: lane n = lane&15, K rows [hi8..hi8+7] and [hi8+16..hi8+23] (row stride ld).
__device__ __forceinline__ v16bf ld_frag_contig16(const bf16* p, int ld) {
  v16bf v;
#pragma unroll
  for (int i = 0; i < 8; ++i) {
    v[i] = p[i * ld];
    v[8 + i] = p[(16 + i) * ld];
  }
  return v;
}

// ---- Generic GEMM: C[M,Nc] = A[M,K](bf16) * Wt[K,Nc](bf16) + epilogue -----
// Block tile 128x128, BK=32, double-buffered LDS staging.
// 8 waves as 4(M) x 2(N); each wave: 32x64 strip = 2x4 WMMA tiles.
// mode 0: Cf = acc + bias
// mode 1: Cf = resid + gamma[n]*(acc + bias)
// mode 2: Cb = gelu(acc + bias)   (bf16 out)
__global__ __launch_bounds__(256)
void gemm_bf16_kernel(const bf16* __restrict__ A, const bf16* __restrict__ Wt,
                      const float* __restrict__ bias, const float* __restrict__ resid,
                      const float* __restrict__ gamma, float* __restrict__ Cf,
                      bf16* __restrict__ Cb, int M, int K, int Nc, int mode)
{
  __shared__ __align__(16) bf16 smA[2][BM * LDA];
  __shared__ __align__(16) bf16 smB[2][BK * LDB];

  const int tid  = threadIdx.x;
  const int lane = tid & 31;
  const int wave = tid >> 5;
  const int n0 = blockIdx.x * BN;
  const int m0 = blockIdx.y * BM;
  const int wm = (wave >> 1) * 32;     // wave row offset inside block tile
  const int wn = (wave & 1) * 64;      // wave col offset inside block tile

  // staging map: per thread 2 A chunks + 2 B chunks (16B each)
  const int arow0 = tid >> 2;                // 0..63
  const int arow1 = arow0 + 64;              // 64..127
  const int apart = (tid & 3) * 8;           // col offset in K-tile (elems)
  const int brow0 = tid >> 4;                // 0..15
  const int brow1 = brow0 + 16;              // 16..31
  const int bpart = (tid & 15) * 8;          // col offset in N-tile (elems)

  int gr0 = m0 + arow0; if (gr0 > M - 1) gr0 = M - 1;
  int gr1 = m0 + arow1; if (gr1 > M - 1) gr1 = M - 1;
  const bf16* A0 = A + (size_t)((int)(gr0 * K)) ;  // K*M < 2^31 for all our GEMMs
  const bf16* A1 = A + (size_t)((int)(gr1 * K));

  v8f acc[2][4] = {};
  v8bf ra0, ra1, rb0, rb1;

  // prologue: stage tile k0=0 into buffer 0
  ra0 = *(const v8bf*)(A0 + apart);
  ra1 = *(const v8bf*)(A1 + apart);
  rb0 = *(const v8bf*)(Wt + (size_t)(brow0 * Nc) + n0 + bpart);
  rb1 = *(const v8bf*)(Wt + (size_t)(brow1 * Nc) + n0 + bpart);
  *(v8bf*)&smA[0][arow0 * LDA + apart] = ra0;
  *(v8bf*)&smA[0][arow1 * LDA + apart] = ra1;
  *(v8bf*)&smB[0][brow0 * LDB + bpart] = rb0;
  *(v8bf*)&smB[0][brow1 * LDB + bpart] = rb1;
  __syncthreads();

  const int a_lrow = wm + (lane & 15);
  const int hi8 = (lane >> 4) << 3;

  for (int k0 = 0; k0 < K; k0 += BK) {
    const int cur = (k0 >> 5) & 1;
    const int nxt = cur ^ 1;
    const bool more = (k0 + BK) < K;
    if (more) {
      const int kn = k0 + BK;
      ra0 = *(const v8bf*)(A0 + kn + apart);
      ra1 = *(const v8bf*)(A1 + kn + apart);
      rb0 = *(const v8bf*)(Wt + (size_t)((kn + brow0) * Nc) + n0 + bpart);
      rb1 = *(const v8bf*)(Wt + (size_t)((kn + brow1) * Nc) + n0 + bpart);
    }
    // fragments from LDS (current buffer)
    v16bf a[2], b[4];
#pragma unroll
    for (int s = 0; s < 2; ++s) {
      const bf16* p = &smA[cur][(a_lrow + s * 16) * LDA + hi8];
      a[s] = ld_frag_rowK(p);
    }
#pragma unroll
    for (int t = 0; t < 4; ++t) {
      const bf16* p = &smB[cur][hi8 * LDB + wn + t * 16 + (lane & 15)];
      b[t] = ld_frag_contig16(p, LDB);
    }
#pragma unroll
    for (int s = 0; s < 2; ++s)
#pragma unroll
      for (int t = 0; t < 4; ++t)
        acc[s][t] = wmma_bf16(a[s], b[t], acc[s][t]);
    __syncthreads();
    if (more) {
      *(v8bf*)&smA[nxt][arow0 * LDA + apart] = ra0;
      *(v8bf*)&smA[nxt][arow1 * LDA + apart] = ra1;
      *(v8bf*)&smB[nxt][brow0 * LDB + bpart] = rb0;
      *(v8bf*)&smB[nxt][brow1 * LDB + bpart] = rb1;
    }
    __syncthreads();
  }

  // epilogue
  const int nlo = lane & 15;
#pragma unroll
  for (int s = 0; s < 2; ++s) {
    const int mbase = m0 + wm + s * 16 + ((lane >> 4) << 3);
#pragma unroll
    for (int t = 0; t < 4; ++t) {
      const int n = n0 + wn + t * 16 + nlo;
      float bv = bias ? bias[n] : 0.0f;
      float gv = gamma ? gamma[n] : 0.0f;
#pragma unroll
      for (int r = 0; r < 8; ++r) {
        const int m = mbase + r;
        if (m >= M) continue;
        const size_t o = (size_t)m * Nc + n;
        float v = acc[s][t][r] + bv;
        if (mode == 0) {
          Cf[o] = v;
        } else if (mode == 1) {
          Cf[o] = resid[o] + gv * v;
        } else {
          float g = 0.5f * v * (1.0f + erff(v * 0.70710678118f));  // exact GELU
          Cb[o] = (bf16)g;
        }
      }
    }
  }
}

// ---- Attention: S = Q * K^T + rel-pos bias --------------------------------
// Q: [BH, NP, HD] bf16 (pre-scaled), Kt: [BH, HD, NP] bf16, S: [BH, NP, NP] f32
__global__ __launch_bounds__(256)
void attn_qk_kernel(const bf16* __restrict__ Qb, const bf16* __restrict__ Ktb,
                    const float* __restrict__ rpb, float* __restrict__ S)
{
  const int lane = threadIdx.x & 31;
  const int wave = threadIdx.x >> 5;
  const int bh = blockIdx.z;
  const int h = bh % Hh;
  const int n0 = blockIdx.x * 32;
  const int m0 = (blockIdx.y * 8 + wave) * 16;

  int rowA = m0 + (lane & 15);
  if (rowA > NP - 1) rowA = NP - 1;
  const int hi8 = (lane >> 4) << 3;
  const bf16* Ap = Qb + ((size_t)bh * NP + rowA) * HDm + hi8;
  const bf16* Kb = Ktb + (size_t)bh * HDm * NP;

  v8f acc[2] = {};
  for (int k0 = 0; k0 < HDm; k0 += 32) {
    v16bf a = ld_frag_rowK(Ap + k0);
    const bf16* Bp = Kb + (k0 + hi8) * NP + n0 + (lane & 15);
    v16bf b0 = ld_frag_contig16(Bp, NP);
    v16bf b1 = ld_frag_contig16(Bp + 16, NP);
    acc[0] = wmma_bf16(a, b0, acc[0]);
    acc[1] = wmma_bf16(a, b1, acc[1]);
  }

  const int nlo = lane & 15;
  const int mbase = m0 + ((lane >> 4) << 3);
  float* Sp = S + (size_t)bh * NP * NP;
#pragma unroll
  for (int t = 0; t < 2; ++t) {
    int j = n0 + 16 * t + nlo;
#pragma unroll
    for (int r = 0; r < 8; ++r) {
      int i = mbase + r;
      if (i >= NP) continue;
      float bias = 0.0f;
      if (i < Nn && j < Nn) {
        int idx;
        if (i == 0 && j == 0)      idx = NRDm - 1;
        else if (j == 0)           idx = NRDm - 2;
        else if (i == 0)           idx = NRDm - 3;
        else {
          int pi = i - 1, pj = j - 1;
          int ri = pi / 14, ci = pi - ri * 14;
          int rj = pj / 14, cj = pj - rj * 14;
          idx = (ri - rj + 13) * 27 + (ci - cj + 13);
        }
        bias = rpb[idx * Hh + h];
      }
      Sp[i * NP + j] = acc[t][r] + bias;
    }
  }
}

// ---- Attention: O = P * V  -> scatter into [B*Nn, D] bf16 at head offset --
__global__ __launch_bounds__(256)
void attn_ov_kernel(const bf16* __restrict__ Pm, const bf16* __restrict__ Vb,
                    bf16* __restrict__ obf)
{
  const int lane = threadIdx.x & 31;
  const int wave = threadIdx.x >> 5;
  const int bh = blockIdx.z;
  const int b = bh / Hh, h = bh % Hh;
  const int n0 = blockIdx.x * 32;
  const int m0 = (blockIdx.y * 8 + wave) * 16;

  int rowA = m0 + (lane & 15);
  if (rowA > NP - 1) rowA = NP - 1;
  const int hi8 = (lane >> 4) << 3;
  const bf16* Ap = Pm + ((size_t)bh * NP + rowA) * NP + hi8;
  const bf16* Vp = Vb + (size_t)bh * NP * HDm;

  v8f acc[2] = {};
  for (int k0 = 0; k0 < NP; k0 += 32) {
    v16bf a = ld_frag_rowK(Ap + k0);
    const bf16* Bp = Vp + (k0 + hi8) * HDm + n0 + (lane & 15);
    v16bf b0 = ld_frag_contig16(Bp, HDm);
    v16bf b1 = ld_frag_contig16(Bp + 16, HDm);
    acc[0] = wmma_bf16(a, b0, acc[0]);
    acc[1] = wmma_bf16(a, b1, acc[1]);
  }

  const int nlo = lane & 15;
  const int mbase = m0 + ((lane >> 4) << 3);
#pragma unroll
  for (int t = 0; t < 2; ++t) {
    int d = n0 + 16 * t + nlo;
#pragma unroll
    for (int r = 0; r < 8; ++r) {
      int i = mbase + r;
      if (i >= Nn) continue;
      obf[(b * Nn + i) * Dd + h * HDm + d] = (bf16)acc[t][r];
    }
  }
}

// ---- QKV split/pack: fp32 [B*Nn, 3D] -> Q(scaled,+bias), K^T, V(+bias) bf16
__global__ void pack_qkv_kernel(const float* __restrict__ qkv,
                                const float* __restrict__ qb,
                                const float* __restrict__ vb,
                                bf16* __restrict__ Qb, bf16* __restrict__ Kt,
                                bf16* __restrict__ Vb)
{
  int idx = blockIdx.x * 256 + threadIdx.x;
  if (idx >= Bb * Hh * NP * HDm) return;
  int d = idx & 63;
  int i = (idx >> 6) % NP;
  int bh = idx / (NP * HDm);
  int b = bh / Hh, h = bh % Hh;
  float q = 0.0f, k = 0.0f, v = 0.0f;
  if (i < Nn) {
    const float* row = qkv + (size_t)(b * Nn + i) * (3 * Dd);
    int c = h * HDm + d;
    q = (row[c] + qb[c]) * SCL;
    k = row[Dd + c];
    v = row[2 * Dd + c] + vb[c];
  }
  Qb[(bh * NP + i) * HDm + d] = (bf16)q;
  Vb[(bh * NP + i) * HDm + d] = (bf16)v;
  Kt[(bh * HDm + d) * NP + i] = (bf16)k;
}

// ---- Row softmax over 197 valid columns, bf16 out, padded cols zeroed -----
__global__ __launch_bounds__(256)
void softmax_kernel(const float* __restrict__ S, bf16* __restrict__ P)
{
  int bh = blockIdx.x;
  int i = blockIdx.y;
  int tid = threadIdx.x;
  const float* row = S + ((size_t)bh * NP + i) * NP;
  bf16* out = P + ((size_t)bh * NP + i) * NP;
  if (i >= Nn) {
    for (int j = tid; j < NP; j += 256) out[j] = (bf16)0.0f;
    return;
  }
  __shared__ float red[8];
  float m = -3.0e38f;
  for (int j = tid; j < Nn; j += 256) m = fmaxf(m, row[j]);
  for (int o = 16; o > 0; o >>= 1) m = fmaxf(m, __shfl_xor(m, o, 32));
  if ((tid & 31) == 0) red[tid >> 5] = m;
  __syncthreads();
  float rmax = red[0];
  for (int w = 1; w < 8; ++w) rmax = fmaxf(rmax, red[w]);
  __syncthreads();

  float s = 0.0f;
  for (int j = tid; j < Nn; j += 256) s += expf(row[j] - rmax);
  for (int o = 16; o > 0; o >>= 1) s += __shfl_xor(s, o, 32);
  if ((tid & 31) == 0) red[tid >> 5] = s;
  __syncthreads();
  float tot = 0.0f;
  for (int w = 0; w < 8; ++w) tot += red[w];
  float inv = 1.0f / tot;
  for (int j = tid; j < NP; j += 256)
    out[j] = (bf16)((j < Nn) ? expf(row[j] - rmax) * inv : 0.0f);
}

// ---- LayerNorm fp32 row -> bf16 row (one block per 768-wide row) ----------
__global__ __launch_bounds__(256)
void ln_bf16_kernel(const float* __restrict__ X, const float* __restrict__ w,
                    const float* __restrict__ b, bf16* __restrict__ Y)
{
  int row = blockIdx.x;
  int tid = threadIdx.x;
  const float* x = X + (size_t)row * Dd;
  float v0 = x[tid], v1 = x[tid + 256], v2 = x[tid + 512];
  __shared__ float red[8];
  float s = v0 + v1 + v2;
  for (int o = 16; o > 0; o >>= 1) s += __shfl_xor(s, o, 32);
  if ((tid & 31) == 0) red[tid >> 5] = s;
  __syncthreads();
  float tot = 0.0f;
  for (int k = 0; k < 8; ++k) tot += red[k];
  float mu = tot * (1.0f / Dd);
  float d0 = v0 - mu, d1 = v1 - mu, d2 = v2 - mu;
  float sq = d0 * d0 + d1 * d1 + d2 * d2;
  __syncthreads();
  for (int o = 16; o > 0; o >>= 1) sq += __shfl_xor(sq, o, 32);
  if ((tid & 31) == 0) red[tid >> 5] = sq;
  __syncthreads();
  float tot2 = 0.0f;
  for (int k = 0; k < 8; ++k) tot2 += red[k];
  float rs = rsqrtf(tot2 * (1.0f / Dd) + 1e-5f);
  bf16* y = Y + (size_t)row * Dd;
  y[tid]       = (bf16)(d0 * rs * w[tid]       + b[tid]);
  y[tid + 256] = (bf16)(d1 * rs * w[tid + 256] + b[tid + 256]);
  y[tid + 512] = (bf16)(d2 * rs * w[tid + 512] + b[tid + 512]);
}

// ---- Mean-pool patch tokens + final LayerNorm -> d_out fp32 ---------------
__global__ __launch_bounds__(256)
void pool_head_kernel(const float* __restrict__ t, const float* __restrict__ w,
                      const float* __restrict__ bias, float* __restrict__ out)
{
  int b = blockIdx.x, tid = threadIdx.x;
  float f0 = 0, f1 = 0, f2 = 0;
  for (int i = 1; i < Nn; ++i) {
    const float* row = t + (size_t)(b * Nn + i) * Dd;
    f0 += row[tid]; f1 += row[tid + 256]; f2 += row[tid + 512];
  }
  const float inv = 1.0f / 196.0f;
  f0 *= inv; f1 *= inv; f2 *= inv;
  __shared__ float red[8];
  float s = f0 + f1 + f2;
  for (int o = 16; o > 0; o >>= 1) s += __shfl_xor(s, o, 32);
  if ((tid & 31) == 0) red[tid >> 5] = s;
  __syncthreads();
  float tot = 0.0f;
  for (int k = 0; k < 8; ++k) tot += red[k];
  float mu = tot * (1.0f / Dd);
  float d0 = f0 - mu, d1 = f1 - mu, d2 = f2 - mu;
  float sq = d0 * d0 + d1 * d1 + d2 * d2;
  __syncthreads();
  for (int o = 16; o > 0; o >>= 1) sq += __shfl_xor(sq, o, 32);
  if ((tid & 31) == 0) red[tid >> 5] = sq;
  __syncthreads();
  float tot2 = 0.0f;
  for (int k = 0; k < 8; ++k) tot2 += red[k];
  float rs = rsqrtf(tot2 * (1.0f / Dd) + 1e-5f);
  float* o2 = out + b * Dd;
  o2[tid]       = d0 * rs * w[tid]       + bias[tid];
  o2[tid + 256] = d1 * rs * w[tid + 256] + bias[tid + 256];
  o2[tid + 512] = d2 * rs * w[tid + 512] + bias[tid + 512];
}

// ---- im2col patches -> bf16 [B*196, 768] ----------------------------------
__global__ void im2col_kernel(const float* __restrict__ x, bf16* __restrict__ pm)
{
  int idx = blockIdx.x * 256 + threadIdx.x;
  if (idx >= MPp * Dd) return;
  int c = idx % Dd;
  int row = idx / Dd;
  int b = row / 196, p = row % 196;
  int ch = c >> 8, rem = c & 255;
  int dy = rem >> 4, dx = rem & 15;
  int py = p / 14, px = p % 14;
  pm[idx] = (bf16)x[((b * 3 + ch) * 224 + (py * 16 + dy)) * 224 + (px * 16 + dx)];
}

// ---- cls token + pos embed assembly -> t fp32 [B, 197, 768] ---------------
__global__ void assemble_kernel(const float* __restrict__ pe, const float* __restrict__ cls,
                                const float* __restrict__ pos, float* __restrict__ t)
{
  int idx = blockIdx.x * 256 + threadIdx.x;
  if (idx >= Mrows * Dd) return;
  int d = idx % Dd;
  int row = idx / Dd;
  int b = row / Nn, i = row % Nn;
  float v = (i == 0) ? cls[d] : pe[(b * 196 + (i - 1)) * Dd + d];
  t[idx] = v + pos[i * Dd + d];
}

// ---- weight transpose+convert: W[Nc,K] f32 -> Wt[K,Nc] bf16 ---------------
__global__ void transw_kernel(const float* __restrict__ W, bf16* __restrict__ Wt,
                              int Nc, int K)
{
  int idx = blockIdx.x * 256 + threadIdx.x;
  if (idx >= Nc * K) return;
  int n = idx % Nc;
  int k = idx / Nc;
  Wt[idx] = (bf16)W[(size_t)n * K + k];
}

// ---------------------------------------------------------------------------
extern "C" void kernel_launch(void* const* d_in, const int* in_sizes, int n_in,
                              void* d_out, int out_size, void* d_ws, size_t ws_size,
                              hipStream_t stream)
{
  (void)in_sizes; (void)n_in; (void)out_size; (void)ws_size;
  const float* x       = (const float*)d_in[0];
  const float* patch_w = (const float*)d_in[1];
  const float* patch_b = (const float*)d_in[2];
  const float* cls_tok = (const float*)d_in[3];
  const float* pos_emb = (const float*)d_in[4];
  const float* ln1_w   = (const float*)d_in[5];
  const float* ln1_b   = (const float*)d_in[6];
  const float* qkv_w   = (const float*)d_in[7];
  const float* q_bias  = (const float*)d_in[8];
  const float* v_bias  = (const float*)d_in[9];
  const float* proj_w  = (const float*)d_in[10];
  const float* proj_b  = (const float*)d_in[11];
  const float* ln2_w   = (const float*)d_in[12];
  const float* ln2_b   = (const float*)d_in[13];
  const float* fc1_w   = (const float*)d_in[14];
  const float* fc1_b   = (const float*)d_in[15];
  const float* fc2_w   = (const float*)d_in[16];
  const float* fc2_b   = (const float*)d_in[17];
  const float* gamma1  = (const float*)d_in[18];
  const float* gamma2  = (const float*)d_in[19];
  const float* rpb     = (const float*)d_in[20];
  const float* fcn_w   = (const float*)d_in[21];
  const float* fcn_b   = (const float*)d_in[22];

  char* base = (char*)d_ws;
  size_t off = 0;
  auto alloc = [&](size_t bytes) -> void* {
    void* p = base + off;
    off = (off + bytes + 255) & ~(size_t)255;
    return p;
  };
  bf16*  pm    = (bf16*)alloc((size_t)MPp * Dd * 2);
  bf16*  pwT   = (bf16*)alloc((size_t)Dd * Dd * 2);
  bf16*  qkvT  = (bf16*)alloc((size_t)Ll * Dd * 3 * Dd * 2);
  bf16*  projT = (bf16*)alloc((size_t)Ll * Dd * Dd * 2);
  bf16*  fc1T  = (bf16*)alloc((size_t)Ll * Dd * FFm * 2);
  bf16*  fc2T  = (bf16*)alloc((size_t)Ll * FFm * Dd * 2);
  float* pe    = (float*)alloc((size_t)MPp * Dd * 4);
  float* t     = (float*)alloc((size_t)Mrows * Dd * 4);
  bf16*  hbf   = (bf16*)alloc((size_t)Mrows * Dd * 2);
  float* qkv   = (float*)alloc((size_t)Mrows * 3 * Dd * 4);
  bf16*  Qb    = (bf16*)alloc((size_t)Bb * Hh * NP * HDm * 2);
  bf16*  Ktb   = (bf16*)alloc((size_t)Bb * Hh * HDm * NP * 2);
  bf16*  Vbm   = (bf16*)alloc((size_t)Bb * Hh * NP * HDm * 2);
  float* S     = (float*)alloc((size_t)Bb * Hh * NP * NP * 4);
  bf16*  Pm    = (bf16*)alloc((size_t)Bb * Hh * NP * NP * 2);
  bf16*  obf   = (bf16*)alloc((size_t)Mrows * Dd * 2);
  bf16*  a1    = (bf16*)alloc((size_t)Mrows * FFm * 2);

  dim3 blk(256);
  auto cdiv = [](int a, int b) { return (a + b - 1) / b; };

  // Weight prep: transpose to [K,N] bf16 (per call; deterministic)
  transw_kernel<<<cdiv(Dd * Dd, 256), blk, 0, stream>>>(patch_w, pwT, Dd, Dd);
  for (int l = 0; l < Ll; ++l) {
    transw_kernel<<<cdiv(3 * Dd * Dd, 256), blk, 0, stream>>>(
        qkv_w + (size_t)l * 3 * Dd * Dd, qkvT + (size_t)l * Dd * 3 * Dd, 3 * Dd, Dd);
    transw_kernel<<<cdiv(Dd * Dd, 256), blk, 0, stream>>>(
        proj_w + (size_t)l * Dd * Dd, projT + (size_t)l * Dd * Dd, Dd, Dd);
    transw_kernel<<<cdiv(FFm * Dd, 256), blk, 0, stream>>>(
        fc1_w + (size_t)l * FFm * Dd, fc1T + (size_t)l * Dd * FFm, FFm, Dd);
    transw_kernel<<<cdiv(Dd * FFm, 256), blk, 0, stream>>>(
        fc2_w + (size_t)l * Dd * FFm, fc2T + (size_t)l * FFm * Dd, Dd, FFm);
  }

  // Patch embed
  im2col_kernel<<<cdiv(MPp * Dd, 256), blk, 0, stream>>>(x, pm);
  gemm_bf16_kernel<<<dim3(Dd / BN, cdiv(MPp, BM)), blk, 0, stream>>>(
      pm, pwT, patch_b, nullptr, nullptr, pe, nullptr, MPp, Dd, Dd, 0);
  assemble_kernel<<<cdiv(Mrows * Dd, 256), blk, 0, stream>>>(pe, cls_tok, pos_emb, t);

  for (int l = 0; l < Ll; ++l) {
    ln_bf16_kernel<<<Mrows, blk, 0, stream>>>(t, ln1_w + l * Dd, ln1_b + l * Dd, hbf);
    gemm_bf16_kernel<<<dim3(3 * Dd / BN, cdiv(Mrows, BM)), blk, 0, stream>>>(
        hbf, qkvT + (size_t)l * Dd * 3 * Dd, nullptr, nullptr, nullptr,
        qkv, nullptr, Mrows, Dd, 3 * Dd, 0);
    pack_qkv_kernel<<<cdiv(Bb * Hh * NP * HDm, 256), blk, 0, stream>>>(
        qkv, q_bias + l * Dd, v_bias + l * Dd, Qb, Ktb, Vbm);
    attn_qk_kernel<<<dim3(NP / 32, 2, Bb * Hh), blk, 0, stream>>>(
        Qb, Ktb, rpb + (size_t)l * NRDm * Hh, S);
    softmax_kernel<<<dim3(Bb * Hh, NP), blk, 0, stream>>>(S, Pm);
    attn_ov_kernel<<<dim3(HDm / 32, 2, Bb * Hh), blk, 0, stream>>>(Pm, Vbm, obf);
    gemm_bf16_kernel<<<dim3(Dd / BN, cdiv(Mrows, BM)), blk, 0, stream>>>(
        obf, projT + (size_t)l * Dd * Dd, proj_b + l * Dd, t, gamma1 + l * Dd,
        t, nullptr, Mrows, Dd, Dd, 1);
    ln_bf16_kernel<<<Mrows, blk, 0, stream>>>(t, ln2_w + l * Dd, ln2_b + l * Dd, hbf);
    gemm_bf16_kernel<<<dim3(FFm / BN, cdiv(Mrows, BM)), blk, 0, stream>>>(
        hbf, fc1T + (size_t)l * Dd * FFm, fc1_b + l * FFm, nullptr, nullptr,
        nullptr, a1, Mrows, Dd, FFm, 2);
    gemm_bf16_kernel<<<dim3(Dd / BN, cdiv(Mrows, BM)), blk, 0, stream>>>(
        a1, fc2T + (size_t)l * FFm * Dd, fc2_b + l * Dd, t, gamma2 + l * Dd,
        t, nullptr, Mrows, FFm, Dd, 1);
  }

  pool_head_kernel<<<Bb, blk, 0, stream>>>(t, fcn_w, fcn_b, (float*)d_out);
}
